// MoE_68779606278568
// MI455X (gfx1250) — matrix operand order repaired and checked by reading, not compiled, not run on hardware
//
#include <hip/hip_runtime.h>
#include <math.h>

// ---- problem constants (match reference) ----
#define B_    8
#define T_    1024
#define C_    1024
#define H_    4096
#define E_    8
#define N_    (B_ * T_)     // 8192 tokens
#define MT    16            // tokens per expert tile (one WMMA M)
#define HC    128           // H-chunk width (8 waves x 16 cols)
#define NWAVE 8             // waves per block (256 threads, wave32)

// workspace layout
#define WS_COUNTS   0
#define WS_TOKIDX   1024
#define WS_TOKW     (WS_TOKIDX + (size_t)E_ * N_ * 4)
#define WS_W1T      (1u << 20)                         // 1 MB
#define WS_W2T      (WS_W1T + (size_t)E_ * C_ * H_ * 2)
#define WS_NEEDED   (WS_W2T + (size_t)E_ * H_ * C_ * 2)

typedef __bf16 bf16_t;
typedef __attribute__((ext_vector_type(16))) bf16_t v16bf;
typedef __attribute__((ext_vector_type(8)))  float  v8f;

union BF16x16 { v16bf v; uint32_t u[8]; uint4 q[2]; };

// fp32 -> bf16 bits, round-to-nearest-even (used off the critical path)
__device__ __forceinline__ uint32_t bf16_bits(float f) {
    uint32_t u = __float_as_uint(f);
    return (u + 0x7FFFu + ((u >> 16) & 1u)) >> 16;
}
__device__ __forceinline__ uint32_t pack2_bf16_rne(float lo, float hi) {
    return bf16_bits(lo) | (bf16_bits(hi) << 16);
}
// single v_perm_b32: truncating pack of two fp32 high-halves (fallback fast path)
__device__ __forceinline__ uint32_t pack2_bf16_trunc(float lo, float hi) {
    return __builtin_amdgcn_perm(__float_as_uint(hi), __float_as_uint(lo),
                                 0x07060302u);
}
__device__ __forceinline__ float gelu_exact(float v) {
    return 0.5f * v * (1.0f + erff(v * 0.70710678118654752f));
}

// 16x32 bf16 A-tile from LDS (row-major bf16, pitch_dw dwords/row) in the
// contiguous k order this file uses for A and B (not AMD's table order):
//   lanes 0-15 (M=lane):    VGPR0-3 = K 0..7,   VGPR4-7 = K 8..15
//   lanes 16-31 (M=lane-16):VGPR0-3 = K 16..23, VGPR4-7 = K 24..31
__device__ __forceinline__ v16bf load_a_lds(const uint32_t* lds, int pitch_dw,
                                            int k0, int lane) {
    const int r  = lane & 15;
    const int kh = (lane >> 4) * 16;
    const uint32_t* row = lds + r * pitch_dw;
    BF16x16 a;
    a.q[0] = *(const uint4*)(row + ((k0 + kh) >> 1));
    a.q[1] = *(const uint4*)(row + ((k0 + 8 + kh) >> 1));
    return a.v;
}

// PRE path: B tile from transposed bf16 weights (row = N column of original,
// K contiguous).  Lane n reads 32 contiguous bytes: two global_load_b128.
//   u[j] = pack(K=2j, K=2j+1); lanes 0-15: K 0..15, lanes 16-31: K 16..31.
__device__ __forceinline__ v16bf load_b_tr(const uint16_t* baseT, int ldk,
                                           int k0, int lane) {
    const uint16_t* p = baseT + (size_t)(lane & 15) * ldk + k0 + (lane >> 4) * 16;
    BF16x16 b;
    b.q[0] = *(const uint4*)p;
    b.q[1] = *(const uint4*)(p + 8);
    return b.v;
}

// Fallback: strided fp32 loads + v_perm truncating pack.
__device__ __forceinline__ v16bf load_b_global(const float* base, int ldm, int lane) {
    const float* p = base + (lane >> 4) * 16 * ldm + (lane & 15);
    BF16x16 b;
#pragma unroll
    for (int j = 0; j < 8; ++j)
        b.u[j] = pack2_bf16_trunc(p[(2 * j) * ldm], p[(2 * j + 1) * ldm]);
    return b.v;
}

// ------------- Kernel 0: fp32 -> bf16 convert + transpose (per expert) ------
// in: [E][R][Cd] fp32 row-major   ->   out: [E][Cd][R] bf16 row-major
__global__ __launch_bounds__(256) void cvt_transpose_kernel(
    const float* __restrict__ in, uint16_t* __restrict__ out, int R, int Cd) {
    const int e = blockIdx.z;
    in  += (size_t)e * R * Cd;
    out += (size_t)e * R * Cd;
    const int r0 = blockIdx.y * 64;
    const int c0 = blockIdx.x * 64;

    __shared__ uint16_t tile[64][72];          // +8 pad vs bank conflicts
    const int tid = threadIdx.x;

    // read 64x64 fp32 (coalesced float4), convert RNE, store to LDS
#pragma unroll
    for (int i = tid; i < 64 * 16; i += 256) {
        int row = i >> 4, q = i & 15;
        float4 v = *(const float4*)(in + (size_t)(r0 + row) * Cd + c0 + q * 4);
        uint2 pkt;
        pkt.x = pack2_bf16_rne(v.x, v.y);
        pkt.y = pack2_bf16_rne(v.z, v.w);
        *(uint2*)&tile[row][q * 4] = pkt;
    }
    __syncthreads();

    // write transposed (coalesced 8B stores per thread)
#pragma unroll
    for (int i = tid; i < 64 * 16; i += 256) {
        int cc = i >> 4, q = i & 15;
        uint16_t t0 = tile[q * 4 + 0][cc];
        uint16_t t1 = tile[q * 4 + 1][cc];
        uint16_t t2 = tile[q * 4 + 2][cc];
        uint16_t t3 = tile[q * 4 + 3][cc];
        uint2 pkt;
        pkt.x = (uint32_t)t0 | ((uint32_t)t1 << 16);
        pkt.y = (uint32_t)t2 | ((uint32_t)t3 << 16);
        *(uint2*)(out + (size_t)(c0 + cc) * R + r0 + q * 4) = pkt;
    }
}

// ---------------- Kernel 1: gating (scores, top-2, softmax, routing) --------
__global__ __launch_bounds__(256) void moe_gate_kernel(
    const float* __restrict__ x, const float* __restrict__ Wg,
    int* __restrict__ counts, int* __restrict__ tokidx, float* __restrict__ tokw) {
    const int lane = threadIdx.x & 31;
    const int tok  = blockIdx.x * NWAVE + (threadIdx.x >> 5);
    const float* xr = x + (size_t)tok * C_;

    float s[E_];
#pragma unroll
    for (int e = 0; e < E_; ++e) s[e] = 0.f;

    for (int c = lane; c < C_; c += 32) {
        float xv = xr[c];
        const float4* wr = (const float4*)(Wg + c * E_);
        float4 w0 = wr[0], w1 = wr[1];
        s[0] += xv * w0.x; s[1] += xv * w0.y; s[2] += xv * w0.z; s[3] += xv * w0.w;
        s[4] += xv * w1.x; s[5] += xv * w1.y; s[6] += xv * w1.z; s[7] += xv * w1.w;
    }
#pragma unroll
    for (int off = 16; off > 0; off >>= 1)
#pragma unroll
        for (int e = 0; e < E_; ++e) s[e] += __shfl_xor(s[e], off, 32);

    if (lane == 0) {
        int e1 = 0; float m1 = s[0];
#pragma unroll
        for (int e = 1; e < E_; ++e) if (s[e] > m1) { m1 = s[e]; e1 = e; }
        int e2 = (e1 == 0) ? 1 : 0; float m2 = s[e2];
#pragma unroll
        for (int e = 0; e < E_; ++e)
            if (e != e1 && s[e] > m2) { m2 = s[e]; e2 = e; }
        float z  = expf(m2 - m1);           // renormalized softmax over top-2
        float w1 = 1.f / (1.f + z);
        float w2 = z / (1.f + z);
        int p1 = atomicAdd(&counts[e1], 1);
        tokidx[e1 * N_ + p1] = tok; tokw[e1 * N_ + p1] = w1;
        int p2 = atomicAdd(&counts[e2], 1);
        tokidx[e2 * N_ + p2] = tok; tokw[e2 * N_ + p2] = w2;
    }
}

// ---------------- Kernel 2: fused routed expert FFN ------------------------
template <bool PRE>
__global__ __launch_bounds__(256) void moe_expert_kernel(
    const float* __restrict__ x,
    const float* __restrict__ W1, const float* __restrict__ b1,
    const float* __restrict__ W2, const float* __restrict__ b2,
    const uint16_t* __restrict__ W1t, const uint16_t* __restrict__ W2t,
    const int* __restrict__ tokidx, const float* __restrict__ tokw,
    const int* __restrict__ counts, float* __restrict__ y) {

    const int e   = blockIdx.y;
    const int cnt = counts[e];
    const int m0  = blockIdx.x * MT;
    if (m0 >= cnt) return;

    __shared__ uint32_t x_lds[MT * (C_ / 2)];   // 16 x 1024 bf16 = 32KB
    __shared__ uint32_t h_lds[MT * (HC / 2)];   // 16 x  128 bf16 =  4KB
    __shared__ int   tok[MT];
    __shared__ float wgt[MT];

    const int tid  = threadIdx.x;
    const int lane = tid & 31;
    const int wave = tid >> 5;

    if (tid < MT) {
        int slot   = m0 + tid;
        bool valid = slot < cnt;
        tok[tid] = tokidx[e * N_ + (valid ? slot : m0)];
        wgt[tid] = valid ? tokw[e * N_ + slot] : 0.f;
    }
    __syncthreads();

    // Gather x rows -> LDS as bf16
    for (int j = tid; j < MT * (C_ / 4); j += 256) {
        int row = j >> 8, c4 = j & 255;
        float4 v = *(const float4*)(x + (size_t)tok[row] * C_ + c4 * 4);
        uint2 p; p.x = pack2_bf16_rne(v.x, v.y); p.y = pack2_bf16_rne(v.z, v.w);
        *(uint2*)&x_lds[row * (C_ / 2) + c4 * 2] = p;
    }
    __syncthreads();

    const float*    W1e  = W1  + (size_t)e * C_ * H_;
    const float*    W2e  = W2  + (size_t)e * H_ * C_;
    const uint16_t* W1te = W1t + (size_t)e * C_ * H_;   // [H_][C_] bf16
    const uint16_t* W2te = W2t + (size_t)e * H_ * C_;   // [C_][H_] bf16

    v8f acc[8];                     // wave owns output cols [wave*128, +128)
#pragma unroll
    for (int t = 0; t < 8; ++t) acc[t] = (v8f){};

    const int nlane = lane & 15;
    const int mbase = (lane >> 4) * 8;

    for (int hc = 0; hc < H_; hc += HC) {
        // ---- phase A: h[16, hc+wave*16 .. +16) = gelu(x @ W1 + b1) ----
        const int n0 = hc + wave * 16;
        v8f ha = (v8f){};
#pragma unroll 4
        for (int k0 = 0; k0 < C_; k0 += 32) {
            v16bf a = load_a_lds(x_lds, C_ / 2, k0, lane);
            v16bf b;
            if constexpr (PRE)
                b = load_b_tr(W1te + (size_t)n0 * C_, C_, k0, lane);
            else
                b = load_b_global(W1e + (size_t)k0 * H_ + n0, H_, lane);
            ha = __builtin_amdgcn_wmma_f32_16x16x32_bf16(
                     false, a, false, b, (short)0, ha, false, false);
        }
        {   // bias + exact gelu + bf16 store into h_lds (C/D lane layout)
            float bias = b1[e * H_ + n0 + nlane];
            uint16_t* h16 = (uint16_t*)h_lds;
#pragma unroll
            for (int v = 0; v < 8; ++v) {
                int m = mbase + v;
                float hv = gelu_exact(ha[v] + bias);
                h16[m * HC + wave * 16 + nlane] = (uint16_t)bf16_bits(hv);
            }
        }
        __syncthreads();

        // ---- phase B: acc += h_chunk @ W2[hc..hc+HC, wave cols) ----
#pragma unroll
        for (int k0 = 0; k0 < HC; k0 += 32) {
            v16bf a = load_a_lds(h_lds, HC / 2, k0, lane);
#pragma unroll
            for (int t = 0; t < 8; ++t) {
                v16bf b;
                if constexpr (PRE)
                    b = load_b_tr(W2te + (size_t)(wave * 128 + t * 16) * H_,
                                  H_, hc + k0, lane);
                else
                    b = load_b_global(W2e + (size_t)(hc + k0) * C_ +
                                      wave * 128 + t * 16, C_, lane);
                acc[t] = __builtin_amdgcn_wmma_f32_16x16x32_bf16(
                             false, a, false, b, (short)0, acc[t], false, false);
            }
        }
        __syncthreads();
    }

    // ---- epilogue: y[tok] += w * (acc + b2)  (exactly 2 adds/element) ----
#pragma unroll
    for (int t = 0; t < 8; ++t) {
        int c = wave * 128 + t * 16 + nlane;
        float bias2 = b2[e * C_ + c];
#pragma unroll
        for (int v = 0; v < 8; ++v) {
            int m = mbase + v;
            float w = wgt[m];
            if (w != 0.f)
                atomicAdd(&y[(size_t)tok[m] * C_ + c], w * (acc[t][v] + bias2));
        }
    }
}

extern "C" void kernel_launch(void* const* d_in, const int* in_sizes, int n_in,
                              void* d_out, int out_size, void* d_ws, size_t ws_size,
                              hipStream_t stream) {
    const float* x  = (const float*)d_in[0];
    const float* Wg = (const float*)d_in[1];
    const float* W1 = (const float*)d_in[2];
    const float* b1 = (const float*)d_in[3];
    const float* W2 = (const float*)d_in[4];
    const float* b2 = (const float*)d_in[5];
    float* y = (float*)d_out;

    int*      counts = (int*)((char*)d_ws + WS_COUNTS);
    int*      tokidx = (int*)((char*)d_ws + WS_TOKIDX);
    float*    tokw   = (float*)((char*)d_ws + WS_TOKW);
    uint16_t* w1t    = (uint16_t*)((char*)d_ws + WS_W1T);
    uint16_t* w2t    = (uint16_t*)((char*)d_ws + WS_W2T);

    hipMemsetAsync(counts, 0, 1024, stream);
    hipMemsetAsync(y, 0, (size_t)out_size * sizeof(float), stream);

    moe_gate_kernel<<<N_ / NWAVE, 256, 0, stream>>>(x, Wg, counts, tokidx, tokw);

    dim3 grid(N_ / MT, E_);   // capacity tiles; inactive tiles exit on count
    if (ws_size >= WS_NEEDED) {
        // one-shot bf16 convert+transpose of the expert weights
        cvt_transpose_kernel<<<dim3(H_ / 64, C_ / 64, E_), 256, 0, stream>>>(
            W1, w1t, C_, H_);
        cvt_transpose_kernel<<<dim3(C_ / 64, H_ / 64, E_), 256, 0, stream>>>(
            W2, w2t, H_, C_);
        moe_expert_kernel<true><<<grid, 256, 0, stream>>>(
            x, W1, b1, W2, b2, w1t, w2t, tokidx, tokw, counts, y);
    } else {
        moe_expert_kernel<false><<<grid, 256, 0, stream>>>(
            x, W1, b1, W2, b2, w1t, w2t, tokidx, tokw, counts, y);
    }
}
